// WeightedGCN4_81793357185100
// MI455X (gfx1250) — matrix operand order repaired and checked by reading, not compiled, not run on hardware
//
#include <hip/hip_runtime.h>
#include <hip/hip_fp16.h>

// ---------------------------------------------------------------------------
// WeightedGCN (hetero GraphSAGE, L=2, H=128) for MI455X / gfx1250.
// Dense GEMMs via v_wmma_f32_16x16x32_f16 with pre-packed (transposed,
// 16-col-padded) f16 weights so both A and B fragments load as 2x b128.
// Edge aggregation: wave32-per-edge coalesced gather + global_atomic_add_f32.
// ---------------------------------------------------------------------------

typedef __attribute__((ext_vector_type(16))) _Float16 v16h;
typedef __attribute__((ext_vector_type(8)))  float    v8f;

union HV16 { v16h v; _Float16 h[16]; uint4 q[2]; };
union FV8  { v8f  v; float f[8]; };
union H4   { _Float16 h[4]; uint2 u; };

static constexpr int NCc = 10000, NFc = 8000, Hc = 128, Lc = 2, OUTDc = 134;
static constexpr int OUTPADc = 144;   // 134 padded to 9*16
static constexpr int E_OCCc = 600000, E_ENTc = 600000, E_PWc = 200000;

#define DEVFN __device__ __forceinline__

DEVFN float wave_sum(float x) {
#pragma unroll
  for (int off = 16; off > 0; off >>= 1) x += __shfl_xor(x, off, 32);
  return x;
}
DEVFN float lrelu(float x) { return x > 0.f ? x : 0.01f * x; }

// ------------------------------- GEMM (WMMA) -------------------------------
// out[M,N] = A1[M,K] @ W1[K,N] (+ A2[M,K] @ W2[K,N]) + bias[N]
// Wt* are pre-packed: f16, column-major [Npad,K], padded cols zero-filled.
// One 16x16 tile per wave; 8 waves per block cover 128 output columns.
// Inner loop is branch-free: 2x b128 (A) + 2x b128 (B) + wmma per operand set.
template <int K, bool HAS_A2>
__global__ void gemm_wmma(const _Float16* __restrict__ A1, const _Float16* __restrict__ Wt1,
                          const _Float16* __restrict__ A2, const _Float16* __restrict__ Wt2,
                          const float* __restrict__ bias, float* __restrict__ out,
                          int M, int N) {
  const int lane    = threadIdx.x & 31;
  const int wave    = threadIdx.x >> 5;
  const int rowbase = blockIdx.x * 16;
  const int colbase = (blockIdx.y * 8 + wave) * 16;
  if (colbase >= N) return;            // wave-uniform: EXEC stays all-ones
  const int m   = lane & 15;           // A row within tile / B column within tile
  const int sel = lane >> 4;           // K-half selector (ISA 7.12.2 layouts)

  // A fragment: lane<16 holds K {0..7,16..23}; lane>=16 holds K {8..15,24..31}
  const _Float16* ap1 = A1 + (size_t)(rowbase + m) * K + sel * 8;
  // B fragment: lane<16 holds K {0..7,16..23}; lane>=16 holds K {8..15,24..31}
  const _Float16* bp1 = Wt1 + (size_t)(colbase + m) * K + sel * 8;
  const _Float16* ap2 = HAS_A2 ? A2 + (size_t)(rowbase + m) * K + sel * 8 : nullptr;
  const _Float16* bp2 = HAS_A2 ? Wt2 + (size_t)(colbase + m) * K + sel * 8 : nullptr;

  v8f acc = {};
#pragma unroll
  for (int kb = 0; kb < K; kb += 32) {
    HV16 a1, b1;
    a1.q[0] = *reinterpret_cast<const uint4*>(ap1 + kb);
    a1.q[1] = *reinterpret_cast<const uint4*>(ap1 + kb + 16);
    b1.q[0] = *reinterpret_cast<const uint4*>(bp1 + kb);
    b1.q[1] = *reinterpret_cast<const uint4*>(bp1 + kb + 16);
    acc = __builtin_amdgcn_wmma_f32_16x16x32_f16(false, a1.v, false, b1.v,
                                                 (short)0, acc, false, false);
    if constexpr (HAS_A2) {
      HV16 a2, b2;
      a2.q[0] = *reinterpret_cast<const uint4*>(ap2 + kb);
      a2.q[1] = *reinterpret_cast<const uint4*>(ap2 + kb + 16);
      b2.q[0] = *reinterpret_cast<const uint4*>(bp2 + kb);
      b2.q[1] = *reinterpret_cast<const uint4*>(bp2 + kb + 16);
      acc = __builtin_amdgcn_wmma_f32_16x16x32_f16(false, a2.v, false, b2.v,
                                                   (short)0, acc, false, false);
    }
  }
  // C/D layout: VGPR r -> M=r (lanes 0-15) / M=r+8 (lanes 16-31); N = lane&15
  const int n = colbase + m;
  if (n < N) {
    const int r0 = rowbase + sel * 8;
    FV8 res; res.v = acc;
    const float bv = bias[n];
#pragma unroll
    for (int r = 0; r < 8; ++r)
      out[(size_t)(r0 + r) * N + n] = res.f[r] + bv;
  }
}

// ---------------------- weight pack: [K,N] f32 -> [Npad,K] f16 -------------
// Wt[n*K + k] = W[k*N + n] for n<N, else 0 (zero cols are WMMA no-ops).
__global__ void pack_wt(const float* __restrict__ W, _Float16* __restrict__ Wt,
                        int K, int N, int Npad) {
  int i = blockIdx.x * blockDim.x + threadIdx.x;
  if (i >= Npad * K) return;
  int n = i / K, k = i - n * K;
  Wt[(size_t)n * K + k] = (n < N) ? (_Float16)W[(size_t)k * N + n] : (_Float16)0.f;
}

// --------------------------- elementwise kernels ---------------------------
__global__ void zero_f32(float* __restrict__ p, int n) {
  int i = blockIdx.x * blockDim.x + threadIdx.x;
  if (i < n) p[i] = 0.f;
}
__global__ void f32_to_f16(const float* __restrict__ s, _Float16* __restrict__ d, int n) {
  int i = blockIdx.x * blockDim.x + threadIdx.x;
  if (i < n) d[i] = (_Float16)s[i];
}
__global__ void relu_f32_to_f16(const float* __restrict__ s, _Float16* __restrict__ d, int n) {
  int i = blockIdx.x * blockDim.x + threadIdx.x;
  if (i < n) d[i] = (_Float16)fmaxf(s[i], 0.f);
}
__global__ void broadcast_row(const float* __restrict__ row, float* __restrict__ o,
                              _Float16* __restrict__ o16, int n) {
  int i = blockIdx.x * blockDim.x + threadIdx.x;
  if (i < n) { float v = row[i & (Hc - 1)]; o[i] = v; o16[i] = (_Float16)v; }
}
__global__ void finalize_mean(const float* __restrict__ s, const float* __restrict__ deg,
                              _Float16* __restrict__ hn16, int n) {
  int i = blockIdx.x * blockDim.x + threadIdx.x;
  if (i < n) {
    float d = fmaxf(deg[i >> 7], 1.f);   // H == 128
    hn16[i] = (_Float16)(s[i] / d);
  }
}

// ------------------------------ edge kernels -------------------------------
__global__ void deg_count(const int* __restrict__ dst, int E, float* __restrict__ deg) {
  int e = blockIdx.x * blockDim.x + threadIdx.x;
  if (e < E) unsafeAtomicAdd(deg + dst[e], 1.0f);
}
// one wave32 per edge: coalesced 512B gather of h_src row, 4 fp32 atomics/lane
__global__ void scatter_edges(const float* __restrict__ hsrc, const int* __restrict__ src,
                              const int* __restrict__ dst, const float* __restrict__ w,
                              int E, float* __restrict__ s) {
  int gid  = blockIdx.x * blockDim.x + threadIdx.x;
  int e    = gid >> 5;
  int lane = gid & 31;
  if (e >= E) return;
  if (lane == 0 && e + 2048 < E) {        // stream-ahead hint for edge arrays
    __builtin_prefetch(src + e + 2048, 0, 0);
    __builtin_prefetch(dst + e + 2048, 0, 0);
    __builtin_prefetch(w + e + 2048, 0, 0);
  }
  int   sr = src[e], ds = dst[e];
  float we = w[e];
  float4 v = reinterpret_cast<const float4*>(hsrc + (size_t)sr * Hc)[lane];
  float* p = s + (size_t)ds * Hc + lane * 4;
  unsafeAtomicAdd(p + 0, v.x * we);
  unsafeAtomicAdd(p + 1, v.y * we);
  unsafeAtomicAdd(p + 2, v.z * we);
  unsafeAtomicAdd(p + 3, v.w * we);
}

// ------------------------- row epilogues (LN etc.) -------------------------
// out = LN(relu(raw)) ; also writes f16 shadow. One wave per 128-wide row.
__global__ void relu_ln_rows(const float* __restrict__ raw, const float* __restrict__ g,
                             const float* __restrict__ be, float* __restrict__ of,
                             _Float16* __restrict__ oh, int M) {
  const int lane = threadIdx.x & 31;
  const int row  = blockIdx.x * (blockDim.x >> 5) + (threadIdx.x >> 5);
  if (row >= M) return;
  float4 v = reinterpret_cast<const float4*>(raw + (size_t)row * Hc)[lane];
  float a[4] = {fmaxf(v.x, 0.f), fmaxf(v.y, 0.f), fmaxf(v.z, 0.f), fmaxf(v.w, 0.f)};
  float mean = wave_sum(a[0] + a[1] + a[2] + a[3]) * (1.f / Hc);
  float q = 0.f;
#pragma unroll
  for (int c = 0; c < 4; ++c) { float d = a[c] - mean; q += d * d; }
  float inv = rsqrtf(wave_sum(q) * (1.f / Hc) + 1e-5f);
  int j = lane * 4;
  float4 o; H4 hh;
#pragma unroll
  for (int c = 0; c < 4; ++c) {
    float t = (a[c] - mean) * inv * g[j + c] + be[j + c];
    (&o.x)[c] = t; hh.h[c] = (_Float16)t;
  }
  reinterpret_cast<float4*>(of + (size_t)row * Hc)[lane] = o;
  *reinterpret_cast<uint2*>(oh + (size_t)row * Hc + j) = hh.u;
}

// hc = relu(LN(c_new)); write hc f32+f16 and hist slice of xcat[:, layer*H .. ]
__global__ void cell_epilogue(const float* __restrict__ raw, const float* __restrict__ g,
                              const float* __restrict__ be, float* __restrict__ hc,
                              _Float16* __restrict__ hc16, float* __restrict__ xcat,
                              int layer, int M) {
  const int lane = threadIdx.x & 31;
  const int row  = blockIdx.x * (blockDim.x >> 5) + (threadIdx.x >> 5);
  if (row >= M) return;
  float4 v = reinterpret_cast<const float4*>(raw + (size_t)row * Hc)[lane];
  float a[4] = {v.x, v.y, v.z, v.w};
  float mean = wave_sum(a[0] + a[1] + a[2] + a[3]) * (1.f / Hc);
  float q = 0.f;
#pragma unroll
  for (int c = 0; c < 4; ++c) { float d = a[c] - mean; q += d * d; }
  float inv = rsqrtf(wave_sum(q) * (1.f / Hc) + 1e-5f);
  int j = lane * 4;
  float4 o; H4 hh;
#pragma unroll
  for (int c = 0; c < 4; ++c) {
    float t = fmaxf((a[c] - mean) * inv * g[j + c] + be[j + c], 0.f);
    (&o.x)[c] = t; hh.h[c] = (_Float16)t;
  }
  reinterpret_cast<float4*>(hc + (size_t)row * Hc)[lane] = o;
  *reinterpret_cast<uint2*>(hc16 + (size_t)row * Hc + j) = hh.u;
  reinterpret_cast<float4*>(xcat + (size_t)row * (Hc * Lc) + layer * Hc)[lane] = o;
}

// hf = relu(0.5*LN(leaky(f0)) + 0.5*LN(leaky(f1)))
__global__ void feat_epilogue(const float* __restrict__ f0r, const float* __restrict__ f1r,
                              const float* __restrict__ g1, const float* __restrict__ b1,
                              const float* __restrict__ g2, const float* __restrict__ b2,
                              float* __restrict__ hf, _Float16* __restrict__ hf16, int M) {
  const int lane = threadIdx.x & 31;
  const int row  = blockIdx.x * (blockDim.x >> 5) + (threadIdx.x >> 5);
  if (row >= M) return;
  float4 va = reinterpret_cast<const float4*>(f0r + (size_t)row * Hc)[lane];
  float4 vb = reinterpret_cast<const float4*>(f1r + (size_t)row * Hc)[lane];
  float a[4] = {lrelu(va.x), lrelu(va.y), lrelu(va.z), lrelu(va.w)};
  float b[4] = {lrelu(vb.x), lrelu(vb.y), lrelu(vb.z), lrelu(vb.w)};
  float ma = wave_sum(a[0] + a[1] + a[2] + a[3]) * (1.f / Hc);
  float mb = wave_sum(b[0] + b[1] + b[2] + b[3]) * (1.f / Hc);
  float qa = 0.f, qb = 0.f;
#pragma unroll
  for (int c = 0; c < 4; ++c) {
    float da = a[c] - ma; qa += da * da;
    float db = b[c] - mb; qb += db * db;
  }
  float ia = rsqrtf(wave_sum(qa) * (1.f / Hc) + 1e-5f);
  float ib = rsqrtf(wave_sum(qb) * (1.f / Hc) + 1e-5f);
  int j = lane * 4;
  float4 o; H4 hh;
#pragma unroll
  for (int c = 0; c < 4; ++c) {
    float h1 = (a[c] - ma) * ia * g1[j + c] + b1[j + c];
    float h2 = (b[c] - mb) * ib * g2[j + c] + b2[j + c];
    float t  = fmaxf(0.5f * h1 + 0.5f * h2, 0.f);
    (&o.x)[c] = t; hh.h[c] = (_Float16)t;
  }
  reinterpret_cast<float4*>(hf + (size_t)row * Hc)[lane] = o;
  *reinterpret_cast<uint2*>(hf16 + (size_t)row * Hc + j) = hh.u;
}

// single broadcast cell row: relu(lin) then LN, one 128-thread block
__global__ void cell_input_row(const float* __restrict__ ec, const float* __restrict__ W,
                               const float* __restrict__ bia, const float* __restrict__ g,
                               const float* __restrict__ be, float* __restrict__ rowout) {
  __shared__ float e[Hc];
  __shared__ float t[Hc];
  __shared__ float red[2];
  int tid = threadIdx.x;
  e[tid] = ec[Hc + tid];                 // embed_cell[1]
  __syncthreads();
  float acc = bia[tid];
  for (int k = 0; k < Hc; ++k) acc += e[k] * W[k * Hc + tid];
  acc = fmaxf(acc, 0.f);
  t[tid] = acc; __syncthreads();
  if (tid == 0) { float s = 0.f; for (int k = 0; k < Hc; ++k) s += t[k]; red[0] = s * (1.f / Hc); }
  __syncthreads();
  float m = red[0];
  float d = acc - m;
  t[tid] = d * d; __syncthreads();
  if (tid == 0) { float s = 0.f; for (int k = 0; k < Hc; ++k) s += t[k]; red[1] = s * (1.f / Hc); }
  __syncthreads();
  float inv = rsqrtf(red[1] + 1e-5f);
  rowout[tid] = d * inv * g[tid] + be[tid];
}

// ------------------------------- launch glue -------------------------------
static inline int cdiv(int a, int b) { return (a + b - 1) / b; }

extern "C" void kernel_launch(void* const* d_in, const int* in_sizes, int n_in,
                              void* d_out, int out_size, void* d_ws, size_t ws_size,
                              hipStream_t stream) {
  (void)in_sizes; (void)n_in; (void)out_size; (void)ws_size;
  // --- param pointers: JAX pytree (sorted dict key) flattening of `params`,
  //     then the edge arrays in setup_inputs() dict order. ---
  const float* embed_cell = (const float*)d_in[0];
  const float* embed_feat = (const float*)d_in[1];
  const float* inlin_c_b  = (const float*)d_in[2];
  const float* inlin_c_w  = (const float*)d_in[3];
  const float* inlin_f_b  = (const float*)d_in[4];
  const float* inlin_f_w  = (const float*)d_in[5];
  const float* inln_c_b   = (const float*)d_in[6];
  const float* inln_c_g   = (const float*)d_in[7];
  const float* inln_f_b   = (const float*)d_in[8];
  const float* inln_f_g   = (const float*)d_in[9];
  // per-layer base: 10 + 15*l
  //  +0 entail.b  +1 entail.w_neigh  +2 entail.w_self
  //  +3 ln_cell.b +4 ln_cell.g +5 ln_h1.b +6 ln_h1.g +7 ln_h2.b +8 ln_h2.g
  //  +9 occur.b  +10 occur.w_neigh +11 occur.w_self
  // +12 pathway.b +13 pathway.w_neigh +14 pathway.w_self
  const float* ro0_b = (const float*)d_in[40];
  const float* ro0_w = (const float*)d_in[41];
  const float* ro1_b = (const float*)d_in[42];
  const float* ro1_w = (const float*)d_in[43];
  const int*   src_occur   = (const int*)d_in[44];
  const int*   dst_occur   = (const int*)d_in[45];
  const float* w_occur     = (const float*)d_in[46];
  const int*   src_entail  = (const int*)d_in[47];
  const int*   dst_entail  = (const int*)d_in[48];
  const float* w_entail    = (const float*)d_in[49];
  const int*   src_pathway = (const int*)d_in[50];
  const int*   dst_pathway = (const int*)d_in[51];
  const float* w_pathway   = (const float*)d_in[52];
  float* out = (float*)d_out;

  // --- workspace layout ---
  char* base = (char*)d_ws;
  size_t off = 0;
  auto alloc = [&](size_t bytes) { size_t o = off; off += (bytes + 255) & ~(size_t)255; return o; };
  float*    hf     = (float*)(base + alloc((size_t)NFc * Hc * 4));
  _Float16* hf16   = (_Float16*)(base + alloc((size_t)NFc * Hc * 2));
  float*    hc     = (float*)(base + alloc((size_t)NCc * Hc * 4));
  _Float16* hc16   = (_Float16*)(base + alloc((size_t)NCc * Hc * 2));
  float*    sA     = (float*)(base + alloc((size_t)NCc * Hc * 4));   // entail agg / c_new raw
  float*    sB     = (float*)(base + alloc((size_t)NFc * Hc * 4));   // occur agg / f0 raw
  float*    sC     = (float*)(base + alloc((size_t)NFc * Hc * 4));   // pathway agg / f1 raw
  _Float16* hnA16  = (_Float16*)(base + alloc((size_t)NCc * Hc * 2));
  _Float16* hnB16  = (_Float16*)(base + alloc((size_t)NFc * Hc * 2));
  _Float16* hnC16  = (_Float16*)(base + alloc((size_t)NFc * Hc * 2));
  float*    degE   = (float*)(base + alloc((size_t)NCc * 4));
  float*    degO   = (float*)(base + alloc((size_t)NFc * 4));
  float*    degP   = (float*)(base + alloc((size_t)NFc * 4));
  float*    xcat   = (float*)(base + alloc((size_t)NCc * Hc * Lc * 4));
  _Float16* xcat16 = (_Float16*)(base + alloc((size_t)NCc * Hc * Lc * 2));
  float*    y0raw  = (float*)(base + alloc((size_t)NCc * 256 * 4));
  _Float16* y016   = (_Float16*)(base + alloc((size_t)NCc * 256 * 2));
  _Float16* winf   = (_Float16*)(base + alloc((size_t)Hc * Hc * 2));       // packed [128,128]
  _Float16* wl[Lc][6];
  for (int l = 0; l < Lc; ++l)
    for (int j = 0; j < 6; ++j)
      wl[l][j] = (_Float16*)(base + alloc((size_t)Hc * Hc * 2));           // packed [128,128]
  _Float16* wro0 = (_Float16*)(base + alloc((size_t)256 * 256 * 2));       // packed [256,256]
  _Float16* wro1 = (_Float16*)(base + alloc((size_t)OUTPADc * 256 * 2));   // packed [144,256]
  float*    crow = (float*)(base + alloc((size_t)Hc * 4));

  const int TB = 256;
  auto ew = [&](int n) { return dim3(cdiv(n, TB)); };

  // --- weight packing (f32 [K,N] -> f16 [Npad,K], once per call) ---
  pack_wt<<<ew(Hc * Hc), TB, 0, stream>>>(inlin_f_w, winf, Hc, Hc, Hc);
  for (int l = 0; l < Lc; ++l) {
    int pb = 10 + 15 * l;
    pack_wt<<<ew(Hc * Hc), TB, 0, stream>>>((const float*)d_in[pb + 2],  wl[l][0], Hc, Hc, Hc); // ent self
    pack_wt<<<ew(Hc * Hc), TB, 0, stream>>>((const float*)d_in[pb + 1],  wl[l][1], Hc, Hc, Hc); // ent neigh
    pack_wt<<<ew(Hc * Hc), TB, 0, stream>>>((const float*)d_in[pb + 11], wl[l][2], Hc, Hc, Hc); // occ self
    pack_wt<<<ew(Hc * Hc), TB, 0, stream>>>((const float*)d_in[pb + 10], wl[l][3], Hc, Hc, Hc); // occ neigh
    pack_wt<<<ew(Hc * Hc), TB, 0, stream>>>((const float*)d_in[pb + 14], wl[l][4], Hc, Hc, Hc); // pw self
    pack_wt<<<ew(Hc * Hc), TB, 0, stream>>>((const float*)d_in[pb + 13], wl[l][5], Hc, Hc, Hc); // pw neigh
  }
  pack_wt<<<ew(256 * 256), TB, 0, stream>>>(ro0_w, wro0, 256, 256, 256);
  pack_wt<<<ew(OUTPADc * 256), TB, 0, stream>>>(ro1_w, wro1, 256, OUTDc, OUTPADc);

  // --- input embeddings ---
  f32_to_f16<<<ew(NFc * Hc), TB, 0, stream>>>(embed_feat, hf16, NFc * Hc);
  gemm_wmma<128, false><<<dim3(NFc / 16, 1), TB, 0, stream>>>(hf16, winf, nullptr, nullptr,
                                                              inlin_f_b, sB, NFc, Hc);
  relu_ln_rows<<<dim3(cdiv(NFc, 8)), TB, 0, stream>>>(sB, inln_f_g, inln_f_b, hf, hf16, NFc);

  cell_input_row<<<1, 128, 0, stream>>>(embed_cell, inlin_c_w, inlin_c_b, inln_c_g, inln_c_b, crow);
  broadcast_row<<<ew(NCc * Hc), TB, 0, stream>>>(crow, hc, hc16, NCc * Hc);

  // --- degrees (weight-independent; once) ---
  zero_f32<<<ew(NCc), TB, 0, stream>>>(degE, NCc);
  zero_f32<<<ew(NFc), TB, 0, stream>>>(degO, NFc);
  zero_f32<<<ew(NFc), TB, 0, stream>>>(degP, NFc);
  deg_count<<<ew(E_ENTc), TB, 0, stream>>>(dst_entail, E_ENTc, degE);
  deg_count<<<ew(E_OCCc), TB, 0, stream>>>(dst_occur, E_OCCc, degO);
  deg_count<<<ew(E_PWc), TB, 0, stream>>>(dst_pathway, E_PWc, degP);

  // --- layers ---
  for (int l = 0; l < Lc; ++l) {
    int pb = 10 + 15 * l;
    const float* ent_b  = (const float*)d_in[pb + 0];
    const float* lnc_b  = (const float*)d_in[pb + 3];
    const float* lnc_g  = (const float*)d_in[pb + 4];
    const float* ln1_b  = (const float*)d_in[pb + 5];
    const float* ln1_g  = (const float*)d_in[pb + 6];
    const float* ln2_b  = (const float*)d_in[pb + 7];
    const float* ln2_g  = (const float*)d_in[pb + 8];
    const float* occ_b  = (const float*)d_in[pb + 9];
    const float* pw_b   = (const float*)d_in[pb + 12];

    zero_f32<<<ew(NCc * Hc), TB, 0, stream>>>(sA, NCc * Hc);
    zero_f32<<<ew(NFc * Hc), TB, 0, stream>>>(sB, NFc * Hc);
    zero_f32<<<ew(NFc * Hc), TB, 0, stream>>>(sC, NFc * Hc);

    scatter_edges<<<dim3(E_ENTc / 8), TB, 0, stream>>>(hf, src_entail, dst_entail, w_entail, E_ENTc, sA);
    scatter_edges<<<dim3(E_OCCc / 8), TB, 0, stream>>>(hc, src_occur, dst_occur, w_occur, E_OCCc, sB);
    scatter_edges<<<dim3(E_PWc / 8), TB, 0, stream>>>(hf, src_pathway, dst_pathway, w_pathway, E_PWc, sC);

    finalize_mean<<<ew(NCc * Hc), TB, 0, stream>>>(sA, degE, hnA16, NCc * Hc);
    finalize_mean<<<ew(NFc * Hc), TB, 0, stream>>>(sB, degO, hnB16, NFc * Hc);
    finalize_mean<<<ew(NFc * Hc), TB, 0, stream>>>(sC, degP, hnC16, NFc * Hc);

    // c_new = hc@Wself + hn@Wneigh + b   (writes back over sA/sB/sC, safe)
    gemm_wmma<128, true><<<dim3(NCc / 16, 1), TB, 0, stream>>>(hc16, wl[l][0], hnA16, wl[l][1],
                                                               ent_b, sA, NCc, Hc);
    gemm_wmma<128, true><<<dim3(NFc / 16, 1), TB, 0, stream>>>(hf16, wl[l][2], hnB16, wl[l][3],
                                                               occ_b, sB, NFc, Hc);
    gemm_wmma<128, true><<<dim3(NFc / 16, 1), TB, 0, stream>>>(hf16, wl[l][4], hnC16, wl[l][5],
                                                               pw_b, sC, NFc, Hc);

    cell_epilogue<<<dim3(cdiv(NCc, 8)), TB, 0, stream>>>(sA, lnc_g, lnc_b, hc, hc16, xcat, l, NCc);
    feat_epilogue<<<dim3(cdiv(NFc, 8)), TB, 0, stream>>>(sB, sC, ln1_g, ln1_b, ln2_g, ln2_b,
                                                         hf, hf16, NFc);
  }

  // --- readout ---
  f32_to_f16<<<ew(NCc * 256), TB, 0, stream>>>(xcat, xcat16, NCc * 256);
  gemm_wmma<256, false><<<dim3(NCc / 16, 2), TB, 0, stream>>>(xcat16, wro0, nullptr, nullptr,
                                                              ro0_b, y0raw, NCc, 256);
  relu_f32_to_f16<<<ew(NCc * 256), TB, 0, stream>>>(y0raw, y016, NCc * 256);
  gemm_wmma<256, false><<<dim3(NCc / 16, 2), TB, 0, stream>>>(y016, wro1, nullptr, nullptr,
                                                              ro1_b, out, NCc, OUTDc);
}
